// TransformerBlock_36206574306111
// MI455X (gfx1250) — matrix operand order repaired and checked by reading, not compiled, not run on hardware
//
#include <hip/hip_runtime.h>

// ---------------------------------------------------------------------------
// TransformerBlock with learned memory slots, tuned for MI455X (gfx1250).
// All GEMMs run on v_wmma_f32_16x16x32_bf16 (f32 accumulate), attention is a
// flash-style single pass with online softmax. wave32 layouts per CDNA5 ISA.
// ---------------------------------------------------------------------------

typedef __attribute__((ext_vector_type(16))) __bf16    v16bf;
typedef __attribute__((ext_vector_type(8)))  float     v8f;
typedef __attribute__((ext_vector_type(4)))  unsigned  u32x4;

union Frag { v16bf v; u32x4 q[2]; };

static constexpr int Bc   = 2;
static constexpr int Sc   = 2048;
static constexpr int Dc   = 1024;
static constexpr int Hc   = 16;
static constexpr int DKc  = 64;
static constexpr int Msl  = 4;
static constexpr int SKc  = Sc + Msl;   // 2052 keys
static constexpr int SKP  = 2080;       // padded to 65 * 32
static constexpr int NTOK = Bc * Sc;    // 4096

static __device__ __forceinline__ unsigned short f2bf(float f) {
    unsigned u = __float_as_uint(f);
    // round-to-nearest-even fp32 -> bf16
    unsigned r = (u + 0x7FFFu + ((u >> 16) & 1u)) >> 16;
    return (unsigned short)r;
}

static __device__ __forceinline__ v8f wmma_bf16(v16bf a, v16bf b, v8f c) {
    return __builtin_amdgcn_wmma_f32_16x16x32_bf16(
        /*neg_a=*/false, a, /*neg_b=*/false, b,
        /*c_mod=*/(short)0, c, /*reuse_a=*/false, /*reuse_b=*/false);
}

// ------------------------------ prep kernels -------------------------------

__global__ void cvt_bf16_kernel(const float* __restrict__ in,
                                unsigned short* __restrict__ out, int n) {
    int i = blockIdx.x * blockDim.x + threadIdx.x;
    if (i < n) out[i] = f2bf(in[i]);
}

// W [K][N] f32 (row-major, in->out)  ->  WT [N][K] bf16
__global__ void wtrans_kernel(const float* __restrict__ W,
                              unsigned short* __restrict__ WT, int K, int N) {
    int i = blockIdx.x * blockDim.x + threadIdx.x;
    if (i >= N * K) return;
    int k = i % K, n = i / K;
    WT[i] = f2bf(W[(size_t)k * N + n]);
}

// append memory K/V slots (and zero the padding rows) into head-major buffers
__global__ void fill_mem_kernel(const float* __restrict__ mk,
                                const float* __restrict__ mv,
                                unsigned short* __restrict__ kbuf,
                                unsigned short* __restrict__ vtb) {
    int t = blockIdx.x * blockDim.x + threadIdx.x;   // B*H*32*64
    if (t >= Bc * Hc * 32 * 64) return;
    int d    = t & 63;
    int slot = (t >> 6) & 31;
    int h    = (t >> 11) & 15;
    int b    = t >> 15;
    int s    = Sc + slot;                            // 2048..2079
    float kv = (slot < Msl) ? mk[(size_t)slot * (Hc * DKc) + h * DKc + d] : 0.f;
    float vv = (slot < Msl) ? mv[(size_t)slot * (Hc * DKc) + h * DKc + d] : 0.f;
    kbuf[(((size_t)(b * Hc + h) * SKP) + s) * DKc + d] = f2bf(kv);
    vtb [(((size_t)(b * Hc + h) * DKc) + d) * SKP + s] = f2bf(vv);
}

// ------------------------------- WMMA GEMM ---------------------------------
// C[m][n] = sum_k A[m][k] * BT[n][k] + bias[n]  (+mode-specific epilogue)
// Block = 4 waves; each wave computes a 64x64 tile (4x4 WMMA accumulators).
// modes: 0=Q  1=K  2=V(transposed)  3=Wo(+x residual, f32+bf16 out)
//        4=W1(relu, bf16 out)       5=W2(+xres residual, f32 out)

__global__ __launch_bounds__(128)
void gemm_wmma_kernel(const unsigned short* __restrict__ A,
                      const unsigned short* __restrict__ BT,
                      const float* __restrict__ bias,
                      const float* __restrict__ resid,
                      unsigned short* __restrict__ outb,
                      float* __restrict__ outf,
                      int Msz, int Nsz, int Ksz, int mode) {
    const int lane = threadIdx.x & 31;
    const int wv   = threadIdx.x >> 5;
    const int ml   = lane & 15;       // row within 16-tile (A) / col (B,D)
    const int kh   = lane >> 4;       // lane half
    const int m0   = blockIdx.y * 128 + (wv >> 1) * 64;
    const int n0   = blockIdx.x * 128 + (wv & 1) * 64;

    const v8f vz = {0.f, 0.f, 0.f, 0.f, 0.f, 0.f, 0.f, 0.f};
    v8f acc[4][4];
    for (int i = 0; i < 4; ++i)
        for (int j = 0; j < 4; ++j) acc[i][j] = vz;

    const unsigned short* arow[4];
    const unsigned short* brow[4];
    for (int i = 0; i < 4; ++i) arow[i] = A  + (size_t)(m0 + i * 16 + ml) * Ksz;
    for (int j = 0; j < 4; ++j) brow[j] = BT + (size_t)(n0 + j * 16 + ml) * Ksz;

    for (int k0 = 0; k0 < Ksz; k0 += 32) {
        Frag af[4], bf[4];
        for (int i = 0; i < 4; ++i) {
            // 16-bit A layout: elems 0..7 -> K = kh*8 + j, elems 8..15 -> 16+kh*8+j
            af[i].q[0] = *(const u32x4*)(arow[i] + k0 + kh * 8);
            af[i].q[1] = *(const u32x4*)(arow[i] + k0 + 16 + kh * 8);
        }
        for (int j = 0; j < 4; ++j) {
            // 16-bit B layout: elems 0..7 -> K = kh*8 + j, elems 8..15 -> 16+kh*8+j
            bf[j].q[0] = *(const u32x4*)(brow[j] + k0 + kh * 8);
            bf[j].q[1] = *(const u32x4*)(brow[j] + k0 + 16 + kh * 8);
        }
        if (k0 + 32 < Ksz) {
            __builtin_prefetch(arow[0] + k0 + 32, 0, 1);
            __builtin_prefetch(brow[0] + k0 + 32, 0, 1);
        }
        for (int i = 0; i < 4; ++i)
            for (int j = 0; j < 4; ++j)
                acc[i][j] = wmma_bf16(af[i].v, bf[j].v, acc[i][j]);
    }

    // Epilogue. D layout: VGPR r holds row m = r + 8*kh, column n = ml.
    for (int i = 0; i < 4; ++i) {
        for (int j = 0; j < 4; ++j) {
            for (int r = 0; r < 8; ++r) {
                int m = m0 + i * 16 + 8 * kh + r;
                int n = n0 + j * 16 + ml;
                float val = acc[i][j][r] + bias[n];
                if (mode == 0 || mode == 1) {           // Q / K head-major
                    int b = m >> 11, s = m & 2047, h = n >> 6, d = n & 63;
                    size_t srow = (mode == 0) ? (size_t)Sc : (size_t)SKP;
                    outb[(((size_t)(b * Hc + h) * srow) + s) * DKc + d] = f2bf(val);
                } else if (mode == 2) {                 // V transposed per head
                    int b = m >> 11, s = m & 2047, h = n >> 6, d = n & 63;
                    outb[(((size_t)(b * Hc + h) * DKc) + d) * SKP + s] = f2bf(val);
                } else if (mode == 3) {                 // X = x + attn@Wo
                    val += resid[(size_t)m * Nsz + n];
                    outf[(size_t)m * Nsz + n] = val;
                    outb[(size_t)m * Nsz + n] = f2bf(val);
                } else if (mode == 4) {                 // relu(X@W1 + b1)
                    val = fmaxf(val, 0.f);
                    outb[(size_t)m * Nsz + n] = f2bf(val);
                } else {                                // out = X + h1@W2 + b2
                    outf[(size_t)m * Nsz + n] = resid[(size_t)m * Nsz + n] + val;
                }
            }
        }
    }
}

// --------------------------- flash attention -------------------------------
// One wave per 16-query tile per head. Keys processed 32 at a time so the
// P @ V GEMM uses a full K=32 bf16 WMMA. Online softmax in D-layout; P is
// transposed D->A layout through a 1KB per-wave LDS bounce.

__global__ __launch_bounds__(128)
void flash_attn_kernel(const unsigned short* __restrict__ qb,
                       const unsigned short* __restrict__ kbuf,
                       const unsigned short* __restrict__ vtb,
                       unsigned short* __restrict__ attnb) {
    __shared__ __align__(16) unsigned short lds[4 * 16 * 32];
    const int lane = threadIdx.x & 31;
    const int wv   = threadIdx.x >> 5;
    const int ml   = lane & 15;
    const int kh   = lane >> 4;

    const int gw = blockIdx.x * 4 + wv;       // B*H*(S/16) waves total
    const int qt = gw & 127;
    const int h  = (gw >> 7) & 15;
    const int b  = gw >> 11;

    unsigned short* myl = lds + wv * (16 * 32);

    // Q fragments (A layout), kept for the whole key loop
    const unsigned short* qbase =
        qb + (((size_t)(b * Hc + h) * Sc) + qt * 16) * DKc;
    Frag aq[2];
    for (int c = 0; c < 2; ++c) {
        const unsigned short* qr = qbase + (size_t)ml * DKc + c * 32;
        aq[c].q[0] = *(const u32x4*)(qr + kh * 16);
        aq[c].q[1] = *(const u32x4*)(qr + kh * 16 + 8);
    }

    const v8f vz = {0.f, 0.f, 0.f, 0.f, 0.f, 0.f, 0.f, 0.f};
    v8f O[4] = {vz, vz, vz, vz};
    float rm[8], rl[8];
    for (int r = 0; r < 8; ++r) { rm[r] = -3.0e38f; rl[r] = 0.f; }

    const unsigned short* kbh = kbuf + ((size_t)(b * Hc + h) * SKP) * DKc;
    const unsigned short* vth = vtb  + ((size_t)(b * Hc + h) * DKc) * SKP;

    for (int it = 0; it < SKP / 32; ++it) {
        const int kbase = it * 32;

        // scores: two 16x16 tiles over 32 keys, K-dim = 64 head dims
        v8f s0 = vz, s1 = vz;
        {
            const unsigned short* kr0 = kbh + (size_t)(kbase + ml) * DKc;
            const unsigned short* kr1 = kbh + (size_t)(kbase + 16 + ml) * DKc;
            for (int c = 0; c < 2; ++c) {
                Frag bk;
                bk.q[0] = *(const u32x4*)(kr0 + c * 32 + kh * 16);
                bk.q[1] = *(const u32x4*)(kr0 + c * 32 + kh * 16 + 8);
                s0 = wmma_bf16(aq[c].v, bk.v, s0);
            }
            for (int c = 0; c < 2; ++c) {
                Frag bk;
                bk.q[0] = *(const u32x4*)(kr1 + c * 32 + kh * 16);
                bk.q[1] = *(const u32x4*)(kr1 + c * 32 + kh * 16 + 8);
                s1 = wmma_bf16(aq[c].v, bk.v, s1);
            }
        }

        // scale + mask invalid key columns (>= 2052)
        const bool ok0 = (kbase + ml)      < SKc;
        const bool ok1 = (kbase + 16 + ml) < SKc;
        float f0[8], f1[8], t[8];
        for (int r = 0; r < 8; ++r) {
            f0[r] = ok0 ? s0[r] * 0.125f : -3.0e38f;   // 1/sqrt(64)
            f1[r] = ok1 ? s1[r] * 0.125f : -3.0e38f;
            t[r]  = fmaxf(f0[r], f1[r]);
        }
        // per-row max across the 16 lanes of this half
        for (int off = 1; off < 16; off <<= 1)
            for (int r = 0; r < 8; ++r)
                t[r] = fmaxf(t[r], __shfl_xor(t[r], off, 16));

        float corr[8], p0[8], p1[8], rs[8];
        for (int r = 0; r < 8; ++r) {
            float nm = fmaxf(rm[r], t[r]);
            corr[r] = __expf(rm[r] - nm);
            p0[r]   = __expf(f0[r] - nm);
            p1[r]   = __expf(f1[r] - nm);
            rm[r]   = nm;
            rs[r]   = p0[r] + p1[r];
        }
        for (int off = 1; off < 16; off <<= 1)
            for (int r = 0; r < 8; ++r)
                rs[r] += __shfl_xor(rs[r], off, 16);
        for (int r = 0; r < 8; ++r) rl[r] = rl[r] * corr[r] + rs[r];
        for (int tt = 0; tt < 4; ++tt)
            for (int r = 0; r < 8; ++r) O[tt][r] *= corr[r];

        // transpose P (D layout -> A layout) through LDS, as bf16
        for (int r = 0; r < 8; ++r) {
            myl[(8 * kh + r) * 32 + ml]      = f2bf(p0[r]);
            myl[(8 * kh + r) * 32 + 16 + ml] = f2bf(p1[r]);
        }
        asm volatile("" ::: "memory");     // DS ops are in-order within a wave
        Frag pa;
        pa.q[0] = *(const u32x4*)(myl + ml * 32 + kh * 16);
        pa.q[1] = *(const u32x4*)(myl + ml * 32 + kh * 16 + 8);
        asm volatile("" ::: "memory");

        // O += P @ V  over 4 d-tiles; V^T gives contiguous B-fragment reads
        for (int tt = 0; tt < 4; ++tt) {
            const unsigned short* vr =
                vth + (size_t)(tt * 16 + ml) * SKP + kbase;
            Frag bv;
            bv.q[0] = *(const u32x4*)(vr + kh * 16);
            bv.q[1] = *(const u32x4*)(vr + kh * 16 + 8);
            O[tt] = wmma_bf16(pa.v, bv.v, O[tt]);
        }
    }

    // normalize and emit token-major bf16 for the Wo GEMM
    for (int tt = 0; tt < 4; ++tt) {
        for (int r = 0; r < 8; ++r) {
            int m = qt * 16 + 8 * kh + r;
            float inv = 1.0f / rl[r];
            attnb[((size_t)(b * Sc + m)) * Dc + h * DKc + tt * 16 + ml] =
                f2bf(O[tt][r] * inv);
        }
    }
}

// ------------------------------- launcher ----------------------------------

extern "C" void kernel_launch(void* const* d_in, const int* in_sizes, int n_in,
                              void* d_out, int out_size, void* d_ws, size_t ws_size,
                              hipStream_t stream) {
    const float* x  = (const float*)d_in[0];
    const float* Wq = (const float*)d_in[1];
    const float* bq = (const float*)d_in[2];
    const float* Wk = (const float*)d_in[3];
    const float* bk = (const float*)d_in[4];
    const float* Wv = (const float*)d_in[5];
    const float* bv = (const float*)d_in[6];
    const float* Wo = (const float*)d_in[7];
    const float* bo = (const float*)d_in[8];
    const float* mk = (const float*)d_in[9];
    const float* mv = (const float*)d_in[10];
    const float* W1 = (const float*)d_in[11];
    const float* b1 = (const float*)d_in[12];
    const float* W2 = (const float*)d_in[13];
    const float* b2 = (const float*)d_in[14];
    float* out = (float*)d_out;

    char* ws = (char*)d_ws;
    size_t off = 0;
    auto alloc = [&](size_t bytes) -> char* {
        char* p = ws + off;
        off += (bytes + 255) & ~(size_t)255;
        return p;
    };

    unsigned short* xb  = (unsigned short*)alloc((size_t)NTOK * Dc * 2);
    unsigned short* WqT = (unsigned short*)alloc((size_t)Dc * Dc * 2);
    unsigned short* WkT = (unsigned short*)alloc((size_t)Dc * Dc * 2);
    unsigned short* WvT = (unsigned short*)alloc((size_t)Dc * Dc * 2);
    unsigned short* WoT = (unsigned short*)alloc((size_t)Dc * Dc * 2);
    unsigned short* W1T = (unsigned short*)alloc((size_t)Dc * Dc * 2);
    unsigned short* W2T = (unsigned short*)alloc((size_t)Dc * Dc * 2);
    unsigned short* qbf = (unsigned short*)alloc((size_t)Bc * Hc * Sc  * DKc * 2);
    unsigned short* kbf = (unsigned short*)alloc((size_t)Bc * Hc * SKP * DKc * 2);
    unsigned short* vtb = (unsigned short*)alloc((size_t)Bc * Hc * DKc * SKP * 2);
    unsigned short* atb = (unsigned short*)alloc((size_t)NTOK * Dc * 2);
    float*          xrs = (float*)         alloc((size_t)NTOK * Dc * 4);
    unsigned short* Xb  = (unsigned short*)alloc((size_t)NTOK * Dc * 2);
    unsigned short* h1b = (unsigned short*)alloc((size_t)NTOK * Dc * 2);
    (void)ws_size; (void)in_sizes; (void)n_in; (void)out_size;

    // 1) fp32 -> bf16 conversions
    {
        int n = NTOK * Dc;
        cvt_bf16_kernel<<<(n + 255) / 256, 256, 0, stream>>>(x, xb, n);
        int wn = Dc * Dc;
        wtrans_kernel<<<(wn + 255) / 256, 256, 0, stream>>>(Wq, WqT, Dc, Dc);
        wtrans_kernel<<<(wn + 255) / 256, 256, 0, stream>>>(Wk, WkT, Dc, Dc);
        wtrans_kernel<<<(wn + 255) / 256, 256, 0, stream>>>(Wv, WvT, Dc, Dc);
        wtrans_kernel<<<(wn + 255) / 256, 256, 0, stream>>>(Wo, WoT, Dc, Dc);
        wtrans_kernel<<<(wn + 255) / 256, 256, 0, stream>>>(W1, W1T, Dc, Dc);
        wtrans_kernel<<<(wn + 255) / 256, 256, 0, stream>>>(W2, W2T, Dc, Dc);
        int fn = Bc * Hc * 32 * 64;
        fill_mem_kernel<<<(fn + 255) / 256, 256, 0, stream>>>(mk, mv, kbf, vtb);
    }

    dim3 ggrid(Dc / 128, NTOK / 128);   // (8, 32)
    // 2) Q / K / V projections (head-major bf16 outputs)
    gemm_wmma_kernel<<<ggrid, 128, 0, stream>>>(xb, WqT, bq, nullptr, qbf, nullptr,
                                                NTOK, Dc, Dc, 0);
    gemm_wmma_kernel<<<ggrid, 128, 0, stream>>>(xb, WkT, bk, nullptr, kbf, nullptr,
                                                NTOK, Dc, Dc, 1);
    gemm_wmma_kernel<<<ggrid, 128, 0, stream>>>(xb, WvT, bv, nullptr, vtb, nullptr,
                                                NTOK, Dc, Dc, 2);

    // 3) flash attention: B*H*(S/16) = 4096 waves, 4 per block
    flash_attn_kernel<<<(Bc * Hc * (Sc / 16)) / 4, 128, 0, stream>>>(qbf, kbf, vtb, atb);

    // 4) X = x + attn @ Wo + bo   (f32 residual kept + bf16 copy for FFN)
    gemm_wmma_kernel<<<ggrid, 128, 0, stream>>>(atb, WoT, bo, x, Xb, xrs,
                                                NTOK, Dc, Dc, 3);
    // 5) h1 = relu(X @ W1 + b1)
    gemm_wmma_kernel<<<ggrid, 128, 0, stream>>>(Xb, W1T, b1, nullptr, h1b, nullptr,
                                                NTOK, Dc, Dc, 4);
    // 6) out = X + h1 @ W2 + b2
    gemm_wmma_kernel<<<ggrid, 128, 0, stream>>>(h1b, W2T, b2, xrs, nullptr, out,
                                                NTOK, Dc, Dc, 5);
}
